// RotatedExpertWrapper_59047210385427
// MI455X (gfx1250) — compile-verified, matrix operand order checked by reading, not run on hardware
//
#include <hip/hip_runtime.h>
#include <stdint.h>

// Rotated SwiGLU expert MLP for MI455X (gfx1250, wave32).
//   0) weights fp32 -> bf16 once (H, gate, up, down) into ws
//   1) rot = bf16( x @ H^T )                      [8192,2048]
//   2) act = bf16( silu(rot@gate^T) * rot@up^T )  [8192,5632]
//   3) out = fp32( act @ down^T )                 [8192,2048]
// GEMM core: v_wmma_f32_16x16x32_bf16, 64x64 per-wave tiles, LDS tiles filled
// with global_load_async_to_lds_b128 (ASYNCcnt), double-buffered.

#define TOKENS 8192
#define HIDDEN 2048
#define INTER  5632
#define PITCH  40   // LDS row pitch in bf16 elems: 80B rows (16B-aligned, bank-clean)

typedef __attribute__((ext_vector_type(16))) __bf16   v16bf;
typedef __attribute__((ext_vector_type(8)))  float    v8f;
typedef __attribute__((ext_vector_type(4)))  float    v4f;
typedef __attribute__((ext_vector_type(4)))  uint32_t v4u;
typedef __attribute__((ext_vector_type(2)))  uint32_t v2u;

__device__ __forceinline__ uint32_t pk_bf16x2(float a, float b) {
  union { float f; uint32_t u; } x, y; x.f = a; y.f = b;
  uint32_t lo = (x.u + 0x7FFFu + ((x.u >> 16) & 1u)) >> 16;          // RNE
  uint32_t hi = (y.u + 0x7FFFu + ((y.u >> 16) & 1u)) & 0xFFFF0000u;  // RNE
  return lo | hi;
}
__device__ __forceinline__ uint16_t f32_to_bf16(float a) {
  union { float f; uint32_t u; } x; x.f = a;
  return (uint16_t)((x.u + 0x7FFFu + ((x.u >> 16) & 1u)) >> 16);
}

// ---- gfx1250 async global->LDS copy (16B per lane), tracked by ASYNCcnt ----
__device__ __forceinline__ void async_b128(uint32_t lds_off, const void* g) {
  asm volatile("global_load_async_to_lds_b128 %0, %1, off"
               :: "v"(lds_off), "v"(g) : "memory");
}
template<int N>
__device__ __forceinline__ void wait_asynccnt() {
  asm volatile("s_wait_asynccnt %0" :: "n"(N) : "memory");
}

// Stage a ROWS x 32 bf16 tile via async copies: ROWS/64 instr/wave (uniform).
template<int ROWS>
__device__ __forceinline__ void stage_async(uint16_t* lds, const uint16_t* src,
                                            int ld, int k0) {
  constexpr int PER = ROWS * 4 / 256;  // 16B chunks per thread
  const int t = threadIdx.x;
#pragma unroll
  for (int i = 0; i < PER; ++i) {
    int idx = t + i * 256;
    int row = idx >> 2, c8 = idx & 3;
    uint32_t lo = (uint32_t)(uintptr_t)(lds + row * PITCH + c8 * 8);
    async_b128(lo, src + (long)row * ld + k0 + c8 * 8);
  }
}

// Stage a ROWS x 32 fp32 tile with on-the-fly bf16 pack (GEMM1 A only).
template<int ROWS>
__device__ __forceinline__ void stage_f32(uint16_t* lds, const float* src,
                                          int ld, int k0) {
  constexpr int PER = ROWS * 8 / 256;  // float4 chunks per thread
  const int t = threadIdx.x;
  v4f tmp[PER];
#pragma unroll
  for (int i = 0; i < PER; ++i) {
    int idx = t + i * 256;
    int row = idx >> 3, c4 = idx & 7;
    tmp[i] = *(const v4f*)(src + (long)row * ld + k0 + c4 * 4);
  }
#pragma unroll
  for (int i = 0; i < PER; ++i) {
    int idx = t + i * 256;
    int row = idx >> 3, c4 = idx & 7;
    v2u p; p.x = pk_bf16x2(tmp[i].x, tmp[i].y); p.y = pk_bf16x2(tmp[i].z, tmp[i].w);
    *(v2u*)(lds + row * PITCH + c4 * 4) = p;
  }
}

union FragBits { v16bf v; v4u u4[2]; };

// CDNA5 16-bit A/B fragment: lane l -> row (l&15), K {kh..kh+7}U{16+kh..16+kh+7},
// kh = (l>>4)*8. Rows are 80B apart -> both halves 16B-aligned ds_load_b128.
__device__ __forceinline__ v16bf lds_frag(const uint16_t* tile, int row0, int lane) {
  const int m  = lane & 15;
  const int kh = (lane >> 4) << 3;
  const uint16_t* p = tile + (row0 + m) * PITCH + kh;
  FragBits f;
  f.u4[0] = *(const v4u*)(p);
  f.u4[1] = *(const v4u*)(p + 16);
  return f.v;
}

__device__ __forceinline__ v8f wmma_bf16(v16bf a, v16bf b, v8f c) {
  return __builtin_amdgcn_wmma_f32_16x16x32_bf16(false, a, false, b, (short)0, c,
                                                 false, false);
}

// -------- weight convert: fp32 -> bf16, 4 floats/thread, exact grids --------
__global__ __launch_bounds__(256)
void cvt_bf16(const float* __restrict__ s, uint16_t* __restrict__ d) {
  long i = (long)blockIdx.x * 256 + threadIdx.x;
  v4f v = *(const v4f*)(s + i * 4);
  v2u p; p.x = pk_bf16x2(v.x, v.y); p.y = pk_bf16x2(v.z, v.w);
  *(v2u*)(d + i * 4) = p;
}

// ------------- generic C = A @ B^T, 128x256 block, 8 waves ------------------
// Wave grid 2(M) x 4(N), wave tile 64x64: 16 WMMA per 16 ds_load_b128.
// B always bf16 (async). A fp32 (packed in-kernel) or bf16 (async).
template<bool A_BF16, bool OUT_BF16>
__global__ __launch_bounds__(256)
void gemm_bt(const void* __restrict__ Ap, const uint16_t* __restrict__ B,
             void* __restrict__ Out, int N, int K) {
  __shared__ __align__(16) uint16_t As[2][128 * PITCH];
  __shared__ __align__(16) uint16_t Bs[2][256 * PITCH];
  constexpr int NASY = A_BF16 ? 6 : 4;  // async instr per wave per stage
  const int gm   = blockIdx.y * 128;
  const int gn   = blockIdx.x * 256;
  const int lane = threadIdx.x & 31;
  const int wv   = threadIdx.x >> 5;
  const int wm   = (wv >> 2) * 64;
  const int wn   = (wv & 3) * 64;

  const float*    Af = (const float*)Ap;
  const uint16_t* Ab = (const uint16_t*)Ap;

  const v8f z = {0.f, 0.f, 0.f, 0.f, 0.f, 0.f, 0.f, 0.f};
  v8f acc[4][4];
#pragma unroll
  for (int i = 0; i < 4; ++i)
#pragma unroll
    for (int j = 0; j < 4; ++j) acc[i][j] = z;

  if (A_BF16) stage_async<128>(As[0], Ab + (long)gm * K, K, 0);
  else        stage_f32 <128>(As[0], Af + (long)gm * K, K, 0);
  stage_async<256>(Bs[0], B + (long)gn * K, K, 0);

  int s = 0;
  for (int k = 0; k < K; k += 32) {
    if (k + 32 < K) {
      if (A_BF16) stage_async<128>(As[s ^ 1], Ab + (long)gm * K, K, k + 32);
      else        stage_f32 <128>(As[s ^ 1], Af + (long)gm * K, K, k + 32);
      stage_async<256>(Bs[s ^ 1], B + (long)gn * K, K, k + 32);
      wait_asynccnt<NASY>();   // stage s complete; stage s^1 still in flight
    } else {
      wait_asynccnt<0>();
    }
    __syncthreads();           // publish stage s
    v16bf af[4];
#pragma unroll
    for (int mt = 0; mt < 4; ++mt) af[mt] = lds_frag(As[s], wm + mt * 16, lane);
#pragma unroll
    for (int nt = 0; nt < 4; ++nt) {
      v16bf bf = lds_frag(Bs[s], wn + nt * 16, lane);
#pragma unroll
      for (int mt = 0; mt < 4; ++mt)
        acc[mt][nt] = wmma_bf16(af[mt], bf, acc[mt][nt]);
    }
    __syncthreads();           // reads of stage s done before refill
    s ^= 1;
  }

  const int rofs = (lane >> 4) * 8;  // C/D: VGPR r -> M = r + 8*(lane>>4)
  const int col  = lane & 15;
#pragma unroll
  for (int mt = 0; mt < 4; ++mt)
#pragma unroll
    for (int nt = 0; nt < 4; ++nt)
#pragma unroll
      for (int r = 0; r < 8; ++r) {
        long grow = gm + wm + mt * 16 + rofs + r;
        long gcol = gn + wn + nt * 16 + col;
        float v = acc[mt][nt][r];
        if (OUT_BF16) ((uint16_t*)Out)[grow * N + gcol] = f32_to_bf16(v);
        else          ((float*)Out)[grow * N + gcol]    = v;
      }
}

// ------- fused gate/up: act = bf16( silu(rot@Wg^T) * (rot@Wu^T) ) -----------
// 128x128 block, 8 waves as 4(M) x 2(N), wave tile 32x64 for both g and u:
// 16 WMMA per 20 ds_load_b128, dual accumulators.
__global__ __launch_bounds__(256)
void gateup(const uint16_t* __restrict__ rot, const uint16_t* __restrict__ Wg,
            const uint16_t* __restrict__ Wu, uint16_t* __restrict__ act,
            int N, int K) {
  __shared__ __align__(16) uint16_t As[2][128 * PITCH];
  __shared__ __align__(16) uint16_t Bg[2][128 * PITCH];
  __shared__ __align__(16) uint16_t Bu[2][128 * PITCH];
  const int gm   = blockIdx.y * 128;
  const int gn   = blockIdx.x * 128;
  const int lane = threadIdx.x & 31;
  const int wv   = threadIdx.x >> 5;
  const int wm   = (wv >> 1) * 32;
  const int wn   = (wv & 1) * 64;

  const v8f z = {0.f, 0.f, 0.f, 0.f, 0.f, 0.f, 0.f, 0.f};
  v8f ag[2][4], au[2][4];
#pragma unroll
  for (int i = 0; i < 2; ++i)
#pragma unroll
    for (int j = 0; j < 4; ++j) { ag[i][j] = z; au[i][j] = z; }

  stage_async<128>(As[0], rot + (long)gm * K, K, 0);
  stage_async<128>(Bg[0], Wg + (long)gn * K, K, 0);
  stage_async<128>(Bu[0], Wu + (long)gn * K, K, 0);

  int s = 0;
  for (int k = 0; k < K; k += 32) {
    if (k + 32 < K) {
      stage_async<128>(As[s ^ 1], rot + (long)gm * K, K, k + 32);
      stage_async<128>(Bg[s ^ 1], Wg + (long)gn * K, K, k + 32);
      stage_async<128>(Bu[s ^ 1], Wu + (long)gn * K, K, k + 32);
      wait_asynccnt<6>();      // newest A:2 + Bg:2 + Bu:2 may remain
    } else {
      wait_asynccnt<0>();
    }
    __syncthreads();
    v16bf af0 = lds_frag(As[s], wm,      lane);
    v16bf af1 = lds_frag(As[s], wm + 16, lane);
#pragma unroll
    for (int nt = 0; nt < 4; ++nt) {
      v16bf bg = lds_frag(Bg[s], wn + nt * 16, lane);
      v16bf bu = lds_frag(Bu[s], wn + nt * 16, lane);
      ag[0][nt] = wmma_bf16(af0, bg, ag[0][nt]);
      ag[1][nt] = wmma_bf16(af1, bg, ag[1][nt]);
      au[0][nt] = wmma_bf16(af0, bu, au[0][nt]);
      au[1][nt] = wmma_bf16(af1, bu, au[1][nt]);
    }
    __syncthreads();
    s ^= 1;
  }

  const int rofs = (lane >> 4) * 8;
  const int col  = lane & 15;
#pragma unroll
  for (int mt = 0; mt < 2; ++mt)
#pragma unroll
    for (int nt = 0; nt < 4; ++nt)
#pragma unroll
      for (int r = 0; r < 8; ++r) {
        long grow = gm + wm + mt * 16 + rofs + r;
        long gcol = gn + wn + nt * 16 + col;
        float g = ag[mt][nt][r];
        float u = au[mt][nt][r];
        float sg = g / (1.0f + __expf(-g));  // silu
        act[grow * N + gcol] = f32_to_bf16(sg * u);
      }
}

extern "C" void kernel_launch(void* const* d_in, const int* in_sizes, int n_in,
                              void* d_out, int out_size, void* d_ws, size_t ws_size,
                              hipStream_t stream) {
  (void)in_sizes; (void)n_in; (void)out_size; (void)ws_size;
  const float* x  = (const float*)d_in[0];
  const float* Hm = (const float*)d_in[1];
  const float* Wg = (const float*)d_in[2];
  const float* Wu = (const float*)d_in[3];
  const float* Wd = (const float*)d_in[4];
  float* out = (float*)d_out;

  // ws layout (bf16): Hb | gb | ub | db | rot | act   (~204 MB total)
  uint16_t* Hb  = (uint16_t*)d_ws;
  uint16_t* gb  = Hb  + (size_t)HIDDEN * HIDDEN;
  uint16_t* ub  = gb  + (size_t)INTER * HIDDEN;
  uint16_t* db  = ub  + (size_t)INTER * HIDDEN;
  uint16_t* rot = db  + (size_t)HIDDEN * INTER;
  uint16_t* act = rot + (size_t)TOKENS * HIDDEN;

  dim3 blk(256);
  cvt_bf16<<<(HIDDEN * HIDDEN) / 1024, blk, 0, stream>>>(Hm, Hb);
  cvt_bf16<<<(INTER * HIDDEN) / 1024, blk, 0, stream>>>(Wg, gb);
  cvt_bf16<<<(INTER * HIDDEN) / 1024, blk, 0, stream>>>(Wu, ub);
  cvt_bf16<<<(HIDDEN * INTER) / 1024, blk, 0, stream>>>(Wd, db);

  gemm_bt<false, true><<<dim3(HIDDEN / 256, TOKENS / 128), blk, 0, stream>>>(
      x, Hb, rot, HIDDEN, HIDDEN);
  gateup<<<dim3(INTER / 128, TOKENS / 128), blk, 0, stream>>>(
      rot, gb, ub, act, INTER, HIDDEN);
  gemm_bt<true, false><<<dim3(HIDDEN / 256, TOKENS / 128), blk, 0, stream>>>(
      act, db, out, HIDDEN, INTER);
}
